// GNNWithPrompt_31860067402230
// MI455X (gfx1250) — compile-verified, matrix operand order checked by reading, not run on hardware
//
#include <hip/hip_runtime.h>
#include <hip/hip_bf16.h>

// ---------------- problem constants (match reference) ----------------
#define N0   200000
#define N1_  50000
#define N2_  10000
#define DIN  128
#define DP   64
#define DH   192      // DIN + DP
#define DHID 256
#define E0C  800000
#define E1C  160000

typedef __bf16 bf16;
typedef __attribute__((ext_vector_type(16))) __bf16 v16bf;
typedef __attribute__((ext_vector_type(8)))  __bf16 v8bf;
typedef __attribute__((ext_vector_type(8)))  float  v8f;

// ---------------------------------------------------------------------
// Fragment layout (ISA 7.12.2, 16-bit operands, wave32):
//   A 16x32: lane&15 = row M, lane>>4 = half; element 2v+j holds
//            K = 16*(v>>2) + 8*half + 2*(v&3) + j
//   => per lane the 16 values are two contiguous 8-element (16B) runs:
//            [8*half .. 8*half+8) and [16+8*half .. 16+8*half+8)
// ---------------------------------------------------------------------

__device__ __forceinline__ v16bf load_a_bf16(const bf16* __restrict__ A, int ld, int lane) {
  const int half = lane >> 4, row = lane & 15;
  const bf16* r = A + (size_t)row * ld + 8 * half;
  v8bf lo = *(const v8bf*)(r);           // global_load_b128
  v8bf hi = *(const v8bf*)(r + 16);      // global_load_b128
  v16bf a;
#pragma unroll
  for (int i = 0; i < 8; ++i) { a[i] = lo[i]; a[8 + i] = hi[i]; }
  return a;
}

__device__ __forceinline__ v16bf load_a_f32(const float* __restrict__ A, int ld, int lane) {
  const int half = lane >> 4, row = lane & 15;
  const float* r = A + (size_t)row * ld + 8 * half;
  v8f lo = *(const v8f*)(r);             // 2x global_load_b128
  v8f hi = *(const v8f*)(r + 16);
  v16bf a;
#pragma unroll
  for (int i = 0; i < 8; ++i) { a[i] = (bf16)lo[i]; a[8 + i] = (bf16)hi[i]; }
  return a;
}

// B fragments come from weights PRE-PACKED into fragment-native layout:
// P[((frag)*32 + lane)*16 + i]  -> one aligned 32B load per lane.
__device__ __forceinline__ v16bf load_b_packed(const bf16* __restrict__ P, int frag, int lane) {
  return *(const v16bf*)(P + ((size_t)frag * 32 + lane) * 16);
}

__device__ __forceinline__ v8f wmma_bf16(v16bf a, v16bf b, v8f c) {
  // 8 args: (neg_a, A, neg_b, B, c_mod, C, reuse_a, reuse_b)
  return __builtin_amdgcn_wmma_f32_16x16x32_bf16(false, a, false, b, (short)0, c,
                                                 false, false);
}

// ---------------- utility kernels ----------------
__global__ void k_zero(float* __restrict__ p, int n) {
  int i = blockIdx.x * blockDim.x + threadIdx.x;
  if (i < n) p[i] = 0.0f;
}

// Convert f32 weight [K,N] row-major -> bf16 fragment-packed layout.
// Output index gid = ((kc*(N/16) + nt)*32 + lane)*16 + i
__global__ void k_pack_b(const float* __restrict__ W, bf16* __restrict__ P, int K, int N) {
  const int gid = blockIdx.x * blockDim.x + threadIdx.x;
  if (gid >= K * N) return;
  const int i    = gid & 15;
  const int lane = (gid >> 4) & 31;
  const int rest = gid >> 9;              // kc*(N/16) + nt
  const int NT   = N >> 4;
  const int nt   = rest % NT;
  const int kc   = rest / NT;
  const int half = lane >> 4, col = lane & 15;
  const int v = i >> 1, j = i & 1;
  const int k = kc * 32 + ((v >> 2) << 4) + (half << 3) + ((v & 3) << 1) + j;
  P[gid] = (bf16)W[(size_t)k * N + nt * 16 + col];
}

// ---------------- prompt MLPs + concat -> h (bf16 [N0, 192]) ----------------
__global__ void k_prompt(const float* __restrict__ feat,
                         const unsigned char* __restrict__ mask,   // jnp bool (1B)
                         const bf16* __restrict__ wpinP, const float* __restrict__ bpin,
                         const bf16* __restrict__ wpoutP, const float* __restrict__ bpout,
                         bf16* __restrict__ hB) {
  const int lane = threadIdx.x & 31;
  const int wave = threadIdx.x >> 5;
  const int tile = blockIdx.x * (blockDim.x >> 5) + wave;
  if (tile >= N0 / 16) return;                 // wave-uniform
  const int row0 = tile * 16;

  // features -> h[:, 0:128] as bf16, 8 elements (16B out / 32B in) per op
  for (int i = lane; i < 16 * (DIN / 8); i += 32) {   // 256 chunks
    const int r = i >> 4, c8 = (i & 15) * 8;
    v8f f = *(const v8f*)(feat + (size_t)(row0 + r) * DIN + c8);
    v8bf o;
#pragma unroll
    for (int j = 0; j < 8; ++j) o[j] = (bf16)f[j];
    *(v8bf*)(hB + (size_t)(row0 + r) * DH + c8) = o;
  }

  // A fragments: 16 rows x 128 K = 4 chunks, f32 -> bf16 in regs
  v16bf a[4];
#pragma unroll
  for (int k = 0; k < 4; ++k)
    a[k] = load_a_f32(feat + (size_t)row0 * DIN + k * 32, DIN, lane);

  const int half = lane >> 4, col16 = lane & 15;
#pragma unroll
  for (int nt = 0; nt < 4; ++nt) {             // 64 output cols = 4 N-tiles
    v8f ci = {}, co = {};
#pragma unroll
    for (int k = 0; k < 4; ++k) {
      ci = wmma_bf16(a[k], load_b_packed(wpinP,  k * 4 + nt, lane), ci);
      co = wmma_bf16(a[k], load_b_packed(wpoutP, k * 4 + nt, lane), co);
    }
    const int col = nt * 16 + col16;
    const float bi = bpin[col], bo = bpout[col];
#pragma unroll
    for (int v = 0; v < 8; ++v) {
      const int row = row0 + v + 8 * half;
      float pi = ci[v] + bi; pi = pi > 0.f ? pi : 0.f;
      float po = co[v] + bo; po = po > 0.f ? po : 0.f;
      hB[(size_t)row * DH + DIN + col] = (bf16)(mask[row] ? pi : po);
    }
  }
}

// ---------------- edge mean-aggregation (gather bf16, f32 atomics) ----------------
// One wave per edge; lane owns an 8-wide contiguous feature chunk (16B load).
__global__ void k_edge_agg(const bf16* __restrict__ h, int F,
                           const int* __restrict__ src, const int* __restrict__ dst, int E,
                           float* __restrict__ agg, float* __restrict__ deg) {
  const int gid  = blockIdx.x * blockDim.x + threadIdx.x;
  const int e    = gid >> 5;
  const int lane = gid & 31;
  if (e >= E) return;
  const int s = src[e], d = dst[e];
  if (e + 1 < E)                               // hide next random-row gather
    __builtin_prefetch(&h[(size_t)src[e + 1] * F], 0, 1);
  if (lane == 0) atomicAdd(deg + d, 1.0f);
  const int nchunk = F >> 3;                   // 24 (F=192) or 32 (F=256)
  for (int c = lane; c < nchunk; c += 32) {
    v8bf x = *(const v8bf*)(h + (size_t)s * F + c * 8);   // global_load_b128
    float* ar = agg + (size_t)d * F + c * 8;
#pragma unroll
    for (int j = 0; j < 8; ++j)
      atomicAdd(ar + j, (float)x[j]);          // no-return atomic, L2-resident
  }
}

// agg/max(deg,1) -> bf16
__global__ void k_norm(const float* __restrict__ agg, const float* __restrict__ deg,
                       bf16* __restrict__ out, int N, int F) {
  const int i = blockIdx.x * blockDim.x + threadIdx.x;   // one 8-chunk per thread
  if (i >= (N * F) >> 3) return;
  const int n = (i << 3) / F;
  float dg = deg[n];
  dg = dg > 1.0f ? dg : 1.0f;
  const float inv = 1.0f / dg;
  v8f x = *(const v8f*)(agg + (size_t)i * 8);
  v8bf o;
#pragma unroll
  for (int j = 0; j < 8; ++j) o[j] = (bf16)(x[j] * inv);
  *(v8bf*)(out + (size_t)i * 8) = o;
}

// ---------------- SAGE layer 0: relu(h[:N1]@Wself + aggN@Wneigh + b) ----------------
__global__ void k_sage0(const bf16* __restrict__ hB, const bf16* __restrict__ aggB,
                        const bf16* __restrict__ wsP, const bf16* __restrict__ wnP,
                        const float* __restrict__ bias, bf16* __restrict__ out) {
  const int lane = threadIdx.x & 31;
  const int wave = threadIdx.x >> 5;
  const int tile = blockIdx.x * (blockDim.x >> 5) + wave;
  if (tile >= N1_ / 16) return;                // wave-uniform
  const int row0 = tile * 16;

  v16bf as[6], aa[6];                          // K=192 -> 6 chunks, both operands
#pragma unroll
  for (int k = 0; k < 6; ++k) {
    as[k] = load_a_bf16(hB   + (size_t)row0 * DH + k * 32, DH, lane);
    aa[k] = load_a_bf16(aggB + (size_t)row0 * DH + k * 32, DH, lane);
  }

  const int half = lane >> 4, col16 = lane & 15;
#pragma unroll 1
  for (int nt = 0; nt < 16; ++nt) {            // 256 cols = 16 N-tiles
    v8f c = {};
#pragma unroll
    for (int k = 0; k < 6; ++k) {
      c = wmma_bf16(as[k], load_b_packed(wsP, k * 16 + nt, lane), c);
      c = wmma_bf16(aa[k], load_b_packed(wnP, k * 16 + nt, lane), c);
    }
    const int col = nt * 16 + col16;
    const float b = bias[col];
#pragma unroll
    for (int v = 0; v < 8; ++v) {
      float x = c[v] + b; x = x > 0.f ? x : 0.f;
      out[(size_t)(row0 + v + 8 * half) * DHID + col] = (bf16)x;
    }
  }
}

// ---------------- SAGE layer 1 (no relu) + 256x2 classifier ----------------
__global__ void k_sage1_cls(const bf16* __restrict__ h1B, const bf16* __restrict__ aggB,
                            const bf16* __restrict__ wsP, const bf16* __restrict__ wnP,
                            const float* __restrict__ bias,
                            const float* __restrict__ wcls, const float* __restrict__ bcls,
                            float* __restrict__ out) {
  __shared__ float h2s[2][16][DHID];           // 32 KB: one 16x256 tile per wave
  const int lane = threadIdx.x & 31;
  const int wave = threadIdx.x >> 5;           // blockDim.x == 64 (2 waves)
  const int tile = blockIdx.x * 2 + wave;
  const bool valid = (tile < N2_ / 16);        // wave-uniform

  if (valid) {
    const int row0 = tile * 16;
    v16bf as[8], aa[8];                        // K=256 -> 8 chunks
#pragma unroll
    for (int k = 0; k < 8; ++k) {
      as[k] = load_a_bf16(h1B  + (size_t)row0 * DHID + k * 32, DHID, lane);
      aa[k] = load_a_bf16(aggB + (size_t)row0 * DHID + k * 32, DHID, lane);
    }
    const int half = lane >> 4, col16 = lane & 15;
#pragma unroll 1
    for (int nt = 0; nt < 16; ++nt) {
      v8f c = {};
#pragma unroll
      for (int k = 0; k < 8; ++k) {
        c = wmma_bf16(as[k], load_b_packed(wsP, k * 16 + nt, lane), c);
        c = wmma_bf16(aa[k], load_b_packed(wnP, k * 16 + nt, lane), c);
      }
      const int col = nt * 16 + col16;
      const float b = bias[col];
#pragma unroll
      for (int v = 0; v < 8; ++v)
        h2s[wave][v + 8 * half][col] = c[v] + b;
    }
  }
  __syncthreads();
  if (valid) {
    // 32 lanes -> 16 rows x 2 logits
    const int r = lane >> 1, j = lane & 1;
    float sum = bcls[j];
    for (int k = 0; k < DHID; ++k)
      sum += h2s[wave][r][k] * wcls[k * 2 + j];
    out[(size_t)(tile * 16 + r) * 2 + j] = sum;
  }
}

// ---------------- host launcher ----------------
extern "C" void kernel_launch(void* const* d_in, const int* in_sizes, int n_in,
                              void* d_out, int out_size, void* d_ws, size_t ws_size,
                              hipStream_t stream) {
  (void)in_sizes; (void)n_in; (void)out_size; (void)ws_size;
  const float*         feat     = (const float*)d_in[0];
  const unsigned char* mask     = (const unsigned char*)d_in[1];  // jnp bool (1B)
  const int*           src0     = (const int*)d_in[2];
  const int*           dst0     = (const int*)d_in[3];
  const int*           src1     = (const int*)d_in[4];
  const int*           dst1     = (const int*)d_in[5];
  /* d_in[6] output_nodes_indices: unused by reference */
  const float* w_pin   = (const float*)d_in[7];
  const float* b_pin   = (const float*)d_in[8];
  const float* w_pout  = (const float*)d_in[9];
  const float* b_pout  = (const float*)d_in[10];
  const float* w_self0 = (const float*)d_in[11];
  const float* w_neigh0= (const float*)d_in[12];
  const float* b0      = (const float*)d_in[13];
  const float* w_self1 = (const float*)d_in[14];
  const float* w_neigh1= (const float*)d_in[15];
  const float* b1      = (const float*)d_in[16];
  const float* w_cls   = (const float*)d_in[17];
  const float* b_cls   = (const float*)d_in[18];

  // -------- workspace carve (256B aligned) --------
  char* p = (char*)d_ws;
  auto carve = [&](size_t bytes) -> char* {
    char* r = p; p += (bytes + 255) & ~(size_t)255; return r;
  };
  bf16*  hB    = (bf16*) carve((size_t)N0  * DH   * sizeof(bf16));
  float* agg0  = (float*)carve((size_t)N1_ * DH   * sizeof(float));
  float* deg0  = (float*)carve((size_t)N1_        * sizeof(float));
  bf16*  agg0B = (bf16*) carve((size_t)N1_ * DH   * sizeof(bf16));
  bf16*  h1B   = (bf16*) carve((size_t)N1_ * DHID * sizeof(bf16));
  float* agg1  = (float*)carve((size_t)N2_ * DHID * sizeof(float));
  float* deg1  = (float*)carve((size_t)N2_        * sizeof(float));
  bf16*  agg1B = (bf16*) carve((size_t)N2_ * DHID * sizeof(bf16));
  bf16*  wpinP = (bf16*) carve((size_t)DIN * DP   * sizeof(bf16));
  bf16*  wpoutP= (bf16*) carve((size_t)DIN * DP   * sizeof(bf16));
  bf16*  ws0P  = (bf16*) carve((size_t)DH  * DHID * sizeof(bf16));
  bf16*  wn0P  = (bf16*) carve((size_t)DH  * DHID * sizeof(bf16));
  bf16*  ws1P  = (bf16*) carve((size_t)DHID* DHID * sizeof(bf16));
  bf16*  wn1P  = (bf16*) carve((size_t)DHID* DHID * sizeof(bf16));

  const int T = 256;
  auto nb = [&](long long n) { return (unsigned)((n + T - 1) / T); };

  // -------- weights f32 -> bf16, fragment-packed --------
  k_pack_b<<<nb(DIN * DP),    T, 0, stream>>>(w_pin,    wpinP,  DIN,  DP);
  k_pack_b<<<nb(DIN * DP),    T, 0, stream>>>(w_pout,   wpoutP, DIN,  DP);
  k_pack_b<<<nb(DH * DHID),   T, 0, stream>>>(w_self0,  ws0P,   DH,   DHID);
  k_pack_b<<<nb(DH * DHID),   T, 0, stream>>>(w_neigh0, wn0P,   DH,   DHID);
  k_pack_b<<<nb(DHID * DHID), T, 0, stream>>>(w_self1,  ws1P,   DHID, DHID);
  k_pack_b<<<nb(DHID * DHID), T, 0, stream>>>(w_neigh1, wn1P,   DHID, DHID);

  // -------- zero accumulators (must re-init every call) --------
  k_zero<<<nb((long long)N1_ * DH),   T, 0, stream>>>(agg0, N1_ * DH);
  k_zero<<<nb(N1_),                   T, 0, stream>>>(deg0, N1_);
  k_zero<<<nb((long long)N2_ * DHID), T, 0, stream>>>(agg1, N2_ * DHID);
  k_zero<<<nb(N2_),                   T, 0, stream>>>(deg1, N2_);

  // -------- prompt MLPs + concat: h[N0,192] bf16 --------
  k_prompt<<<(N0 / 16 + 3) / 4, 128, 0, stream>>>(feat, mask, wpinP, b_pin,
                                                  wpoutP, b_pout, hB);

  // -------- layer 0 aggregation --------
  k_edge_agg<<<nb((long long)E0C * 32), T, 0, stream>>>(hB, DH, src0, dst0, E0C,
                                                        agg0, deg0);
  k_norm<<<nb(((long long)N1_ * DH) >> 3), T, 0, stream>>>(agg0, deg0, agg0B, N1_, DH);

  // -------- layer 0 GEMMs --------
  k_sage0<<<(N1_ / 16 + 3) / 4, 128, 0, stream>>>(hB, agg0B, ws0P, wn0P, b0, h1B);

  // -------- layer 1 aggregation --------
  k_edge_agg<<<nb((long long)E1C * 32), T, 0, stream>>>(h1B, DHID, src1, dst1, E1C,
                                                        agg1, deg1);
  k_norm<<<nb(((long long)N2_ * DHID) >> 3), T, 0, stream>>>(agg1, deg1, agg1B, N2_, DHID);

  // -------- layer 1 GEMMs + classifier --------
  k_sage1_cls<<<(N2_ / 16 + 1) / 2, 64, 0, stream>>>(h1B, agg1B, ws1P, wn1P, b1,
                                                     w_cls, b_cls, (float*)d_out);
}
